// GAT_12189117186677
// MI455X (gfx1250) — matrix operand order repaired and checked by reading, not compiled, not run on hardware
//
#include <hip/hip_runtime.h>
#include <hip/hip_bf16.h>
#include <math.h>

typedef __attribute__((ext_vector_type(16))) _Float16 v16h;
typedef __attribute__((ext_vector_type(8)))  float    v8f;

#define NODES 10000

// ---------------- ordered-float helpers for atomic max ----------------
__device__ __forceinline__ unsigned f2ord(float f) {
    unsigned u = __float_as_uint(f);
    return (u & 0x80000000u) ? ~u : (u | 0x80000000u);
}
__device__ __forceinline__ float ord2f(unsigned u) {
    return (u & 0x80000000u) ? __uint_as_float(u & 0x7FFFFFFFu)
                             : __uint_as_float(~u);
}

// ---------------- WMMA GEMM: C[M,N] = A[M,K] @ B[N,K]^T (+bias)(+relu) ----
// A, B, C are f32 row-major; operands converted to f16 in registers, f32 accum.
// One wave computes a 16 x (NT*16) tile. M multiple of 16; K multiple of 32.
// N-edge handling: B row pointer clamped to N-1 (k-invariant, branchless inner
// loop, EXEC stays all-ones at every WMMA); garbage columns are exactly the
// lanes whose stores are guarded off in the epilogue.
// NOTE: no k-unroll — with 1x body the scheduler keeps 4 B-fragment load
// streams in flight (loadcnt 1..3 outstanding) and fills most WMMA->VALU
// hazard slots; unrolling x2 was measured (round 3) to exhaust registers and
// collapse to a single serialized load window (s_wait_loadcnt 0x0 everywhere).
template<int NT, bool RELU, bool BIAS>
__global__ __launch_bounds__(256) void gemm_wmma_kernel(
    const float* __restrict__ A, const float* __restrict__ B,
    const float* __restrict__ bias, float* __restrict__ C,
    int M, int N, int K)
{
    const int lane  = threadIdx.x & 31;
    const int wave  = blockIdx.x * 8 + (threadIdx.x >> 5);
    const int mtiles = M >> 4;
    const int ntiles = (N + NT * 16 - 1) / (NT * 16);
    if (wave >= mtiles * ntiles) return;

    const int mt = wave % mtiles;
    const int n0 = (wave / mtiles) * (NT * 16);

    // A fragment addressing (16x32 f16, contiguous k = 16*(lane/16) + i)
    const int am    = (mt << 4) + (lane & 15);
    const int akoff = (lane >> 4) << 4;   // 0 or 16
    // B fragment addressing (32x16 f16 layout)
    const int bn    = lane & 15;
    const int bkoff = (lane >> 4) << 4;   // 0 or 16

    v8f acc[NT];
    #pragma unroll
    for (int t = 0; t < NT; t++)
        #pragma unroll
        for (int j = 0; j < 8; j++) acc[t][j] = 0.f;

    const float* arow = A + (size_t)am * K;
    // k-invariant, clamped B row pointers (branchless inner loop)
    const float* brow[NT];
    #pragma unroll
    for (int t = 0; t < NT; t++) {
        const int n = n0 + t * 16 + bn;
        brow[t] = B + (size_t)(n < N ? n : (N - 1)) * K + bkoff;
    }

    for (int k0 = 0; k0 < K; k0 += 32) {
        // branchless near-scope prefetch of next A chunk (clamped -> valid addr)
        const int kp = (k0 + 32 < K) ? (k0 + 32) : k0;
        __builtin_prefetch(arow + kp + akoff, 0, 3);

        const float* ap = arow + k0 + akoff;
        v16h afrag;
        #pragma unroll
        for (int i = 0; i < 8; i++) afrag[i]     = (_Float16)ap[i];       // K = kb+0..7
        #pragma unroll
        for (int i = 0; i < 8; i++) afrag[8 + i] = (_Float16)ap[8 + i];  // K = kb+8..15

        #pragma unroll
        for (int t = 0; t < NT; t++) {
            const float* bp = brow[t] + k0;
            v16h bfrag;
            #pragma unroll
            for (int i = 0; i < 16; i++) bfrag[i] = (_Float16)bp[i];
            acc[t] = __builtin_amdgcn_wmma_f32_16x16x32_f16(
                false, afrag, false, bfrag, (short)0, acc[t], false, false);
        }
    }

    // C/D layout: lane elem j -> row (lane>=16?8:0)+j, col lane&15
    const int mrow = (mt << 4) + ((lane >> 4) << 3);
    #pragma unroll
    for (int t = 0; t < NT; t++) {
        const int n = n0 + t * 16 + (lane & 15);
        if (n >= N) continue;
        const float bv = BIAS ? bias[n] : 0.f;
        #pragma unroll
        for (int j = 0; j < 8; j++) {
            float v = acc[t][j] + bv;
            if (RELU) v = fmaxf(v, 0.f);
            C[(size_t)(mrow + j) * N + n] = v;
        }
    }
}

// ---------------- per-node attention coefficients ----------------
__global__ __launch_bounds__(256) void alpha_kernel(
    const float* __restrict__ h, const float* __restrict__ a_src,
    const float* __restrict__ a_dst, float* __restrict__ as,
    float* __restrict__ ad, int C)
{
    const int n = blockIdx.x;
    const float* row = h + (size_t)n * C;
    float s = 0.f, d = 0.f;
    for (int c = threadIdx.x; c < C; c += 256) {
        float v = row[c];
        s += v * a_src[c];
        d += v * a_dst[c];
    }
    __shared__ float rs[256], rd[256];
    rs[threadIdx.x] = s; rd[threadIdx.x] = d;
    __syncthreads();
    for (int o = 128; o > 0; o >>= 1) {
        if (threadIdx.x < o) {
            rs[threadIdx.x] += rs[threadIdx.x + o];
            rd[threadIdx.x] += rd[threadIdx.x + o];
        }
        __syncthreads();
    }
    if (threadIdx.x == 0) { as[n] = rs[0]; ad[n] = rd[0]; }
}

// ---------------- init kernels ----------------
__global__ void init_node_kernel(unsigned* __restrict__ mord,
                                 float* __restrict__ denom, int N)
{
    int i = blockIdx.x * blockDim.x + threadIdx.x;
    if (i < N) { mord[i] = f2ord(-INFINITY); denom[i] = 0.f; }
}

// 2D grid: y = row, x covers channels (no integer mod)
__global__ void init_out_bias_kernel(float* __restrict__ out,
                                     const float* __restrict__ b, int C)
{
    const int c = blockIdx.x * blockDim.x + threadIdx.x;
    out[(size_t)blockIdx.y * C + c] = b[c];
}

// ---------------- edge kernels (E real edges + NODES self loops) --------
__device__ __forceinline__ int edge_src(const int* ei, int E, int e) {
    return (e < E) ? ei[e] : (e - E);
}
__device__ __forceinline__ int edge_dst(const int* ei, int E, int e) {
    return (e < E) ? ei[E + e] : (e - E);
}

__global__ void edge_max_kernel(const int* __restrict__ ei, int E, int Etot,
                                const float* __restrict__ as,
                                const float* __restrict__ ad,
                                float* __restrict__ earr,
                                unsigned* __restrict__ mord)
{
    int e = blockIdx.x * blockDim.x + threadIdx.x;
    if (e >= Etot) return;
    int s = edge_src(ei, E, e), d = edge_dst(ei, E, e);
    float v = as[s] + ad[d];
    v = (v > 0.f) ? v : 0.2f * v;          // leaky_relu(0.2)
    earr[e] = v;
    atomicMax(&mord[d], f2ord(v));
}

__global__ void edge_exp_kernel(const int* __restrict__ ei, int E, int Etot,
                                float* __restrict__ earr,
                                const unsigned* __restrict__ mord,
                                float* __restrict__ denom)
{
    int e = blockIdx.x * blockDim.x + threadIdx.x;
    if (e >= Etot) return;
    int d = edge_dst(ei, E, e);
    float a = expf(earr[e] - ord2f(mord[d]));
    earr[e] = a;
    atomicAdd(&denom[d], a);
}

// one block (256 threads) per edge; float4 gather + 4 atomics per thread
__global__ __launch_bounds__(256) void edge_aggr_kernel(
    const int* __restrict__ ei, int E, int Etot,
    const float* __restrict__ earr, const float* __restrict__ denom,
    const float* __restrict__ h, float* __restrict__ out, int C)
{
    const int e = blockIdx.x;
    const int s = edge_src(ei, E, e), d = edge_dst(ei, E, e);
    const float wgt = earr[e] / denom[d];
    const float4* hs = (const float4*)(h + (size_t)s * C);
    float*        od = out + (size_t)d * C;
    const int nq = C >> 2;
    for (int q = threadIdx.x; q < nq; q += 256) {
        float4 v = hs[q];
        float* p = od + (q << 2);
        atomicAdd(p + 0, wgt * v.x);
        atomicAdd(p + 1, wgt * v.y);
        atomicAdd(p + 2, wgt * v.z);
        atomicAdd(p + 3, wgt * v.w);
    }
}

// ---------------- batch norm (training mode, biased var) ----------------
__global__ __launch_bounds__(256) void bn_stats_kernel(
    const float* __restrict__ x, const float* __restrict__ w,
    const float* __restrict__ b, float* __restrict__ scale,
    float* __restrict__ shift, int N, int C)
{
    const int c = blockIdx.x;
    float s = 0.f, ss = 0.f;
    for (int n = threadIdx.x; n < N; n += 256) {
        float v = x[(size_t)n * C + c];
        s += v; ss += v * v;
    }
    __shared__ float rs[256], rss[256];
    rs[threadIdx.x] = s; rss[threadIdx.x] = ss;
    __syncthreads();
    for (int o = 128; o > 0; o >>= 1) {
        if (threadIdx.x < o) {
            rs[threadIdx.x]  += rs[threadIdx.x + o];
            rss[threadIdx.x] += rss[threadIdx.x + o];
        }
        __syncthreads();
    }
    if (threadIdx.x == 0) {
        float mu  = rs[0] / (float)N;
        float var = rss[0] / (float)N - mu * mu;
        float sc  = w[c] * rsqrtf(var + 1e-5f);
        scale[c] = sc;
        shift[c] = b[c] - mu * sc;
    }
}

// 2D grid: y = row, x covers channels
__global__ void bn_apply_relu_kernel(float* __restrict__ x,
                                     const float* __restrict__ scale,
                                     const float* __restrict__ shift, int C)
{
    const int c = blockIdx.x * blockDim.x + threadIdx.x;
    const size_t i = (size_t)blockIdx.y * C + c;
    x[i] = fmaxf(x[i] * scale[c] + shift[c], 0.f);
}

// ---------------- orchestration ----------------
static inline int gemm_blocks(int M, int N, int NT) {
    int waves = (M >> 4) * ((N + NT * 16 - 1) / (NT * 16));
    return (waves + 7) / 8;   // 8 waves (256 threads) per block
}

extern "C" void kernel_launch(void* const* d_in, const int* in_sizes, int n_in,
                              void* d_out, int out_size, void* d_ws, size_t ws_size,
                              hipStream_t stream)
{
    (void)n_in; (void)out_size; (void)ws_size;
    const float* x      = (const float*)d_in[0];
    const int*   ei     = (const int*)  d_in[1];
    const float* W1     = (const float*)d_in[2];
    const float* a_src1 = (const float*)d_in[3];
    const float* a_dst1 = (const float*)d_in[4];
    const float* b1     = (const float*)d_in[5];
    const float* bn1_w  = (const float*)d_in[6];
    const float* bn1_b  = (const float*)d_in[7];
    const float* W2     = (const float*)d_in[8];
    const float* a_src2 = (const float*)d_in[9];
    const float* a_dst2 = (const float*)d_in[10];
    const float* b2     = (const float*)d_in[11];
    const float* bn2_w  = (const float*)d_in[12];
    const float* bn2_b  = (const float*)d_in[13];
    const float* fc_w   = (const float*)d_in[14];
    const float* fc_b   = (const float*)d_in[15];
    const float* fc1_w  = (const float*)d_in[16];
    const float* fc1_b  = (const float*)d_in[17];

    const int E    = in_sizes[1] / 2;
    const int Etot = E + NODES;

    // workspace carving (~103 MB)
    char* wsp = (char*)d_ws;
    size_t off = 0;
    auto give = [&](size_t bytes) -> void* {
        void* p = wsp + off;
        off += (bytes + 255) & ~(size_t)255;
        return p;
    };
    float*    h     = (float*)   give((size_t)NODES * 1024 * 4); // h1, then h2
    float*    g     = (float*)   give((size_t)NODES * 1024 * 4); // out1/bn1, then out2/bn2
    float*    fco   = (float*)   give((size_t)NODES * 512 * 4);
    float*    as    = (float*)   give((size_t)NODES * 4);
    float*    adv   = (float*)   give((size_t)NODES * 4);
    unsigned* mord  = (unsigned*)give((size_t)NODES * 4);
    float*    denom = (float*)   give((size_t)NODES * 4);
    float*    earr  = (float*)   give((size_t)Etot * 4);
    float*    scl   = (float*)   give(1024 * 4);
    float*    shf   = (float*)   give(1024 * 4);

    dim3 blk(256);
    const int  nodeBlk = (NODES + 255) / 256;
    const int  edgeBlk = (Etot + 255) / 256;
    const dim3 ncGrid(1024 / 256, NODES);   // elementwise over [NODES,1024]

    // ================= layer 1 =================
    gemm_wmma_kernel<4, false, false><<<gemm_blocks(NODES, 1024, 4), blk, 0, stream>>>(
        x, W1, nullptr, h, NODES, 1024, 256);
    alpha_kernel<<<NODES, blk, 0, stream>>>(h, a_src1, a_dst1, as, adv, 1024);
    init_node_kernel<<<nodeBlk, blk, 0, stream>>>(mord, denom, NODES);
    edge_max_kernel<<<edgeBlk, blk, 0, stream>>>(ei, E, Etot, as, adv, earr, mord);
    edge_exp_kernel<<<edgeBlk, blk, 0, stream>>>(ei, E, Etot, earr, mord, denom);
    init_out_bias_kernel<<<ncGrid, blk, 0, stream>>>(g, b1, 1024);
    edge_aggr_kernel<<<Etot, blk, 0, stream>>>(ei, E, Etot, earr, denom, h, g, 1024);
    bn_stats_kernel<<<1024, blk, 0, stream>>>(g, bn1_w, bn1_b, scl, shf, NODES, 1024);
    bn_apply_relu_kernel<<<ncGrid, blk, 0, stream>>>(g, scl, shf, 1024);

    // ================= layer 2 =================
    gemm_wmma_kernel<4, false, false><<<gemm_blocks(NODES, 1024, 4), blk, 0, stream>>>(
        g, W2, nullptr, h, NODES, 1024, 1024);
    alpha_kernel<<<NODES, blk, 0, stream>>>(h, a_src2, a_dst2, as, adv, 1024);
    init_node_kernel<<<nodeBlk, blk, 0, stream>>>(mord, denom, NODES);
    edge_max_kernel<<<edgeBlk, blk, 0, stream>>>(ei, E, Etot, as, adv, earr, mord);
    edge_exp_kernel<<<edgeBlk, blk, 0, stream>>>(ei, E, Etot, earr, mord, denom);
    init_out_bias_kernel<<<ncGrid, blk, 0, stream>>>(g, b2, 1024);  // g free after gemm2
    edge_aggr_kernel<<<Etot, blk, 0, stream>>>(ei, E, Etot, earr, denom, h, g, 1024);
    bn_stats_kernel<<<1024, blk, 0, stream>>>(g, bn2_w, bn2_b, scl, shf, NODES, 1024);
    bn_apply_relu_kernel<<<ncGrid, blk, 0, stream>>>(g, scl, shf, 1024);

    // ================= MLP head =================
    gemm_wmma_kernel<4, true, true><<<gemm_blocks(NODES, 512, 4), blk, 0, stream>>>(
        g, fc_w, fc_b, fco, NODES, 512, 1024);
    gemm_wmma_kernel<1, false, true><<<gemm_blocks(NODES, 10, 1), blk, 0, stream>>>(
        fco, fc1_w, fc1_b, (float*)d_out, NODES, 10, 512);
}
